// RegionAwareAttention_41197326303869
// MI455X (gfx1250) — compile-verified, matrix operand order checked by reading, not run on hardware
//
#include <hip/hip_runtime.h>
#include <cstdint>
#include <cstddef>

// ---------------------------------------------------------------------------
// Types for CDNA5 WMMA (wave32, 16x16x32 bf16 -> f32 accumulate)
// ---------------------------------------------------------------------------
typedef __attribute__((ext_vector_type(16))) __bf16 bf16x16;
typedef __attribute__((ext_vector_type(8)))  float  f32x8;

union FragB { bf16x16 v; uint4 u[2]; };

__device__ __forceinline__ unsigned short f2bf(float f) {
    union { float f; unsigned int u; } c; c.f = f;
    unsigned int u = c.u;
    unsigned int r = (u + 0x7FFFu + ((u >> 16) & 1u)) >> 16;   // round-nearest-even
    return (unsigned short)r;
}

// ---------------------------------------------------------------------------
// f32 -> bf16 conversion (weights)
// ---------------------------------------------------------------------------
__global__ void cvt_f32_bf16(const float* __restrict__ in,
                             unsigned short* __restrict__ out, int n) {
    int i = blockIdx.x * blockDim.x + threadIdx.x;
    if (i < n) out[i] = f2bf(in[i]);
}

// ---------------------------------------------------------------------------
// Generic GEMM: C(MxN) = A(MxK) * B(KxN) + bias, bf16 WMMA, f32 accumulate.
// A: f32 or bf16 (template). B: bf16 (preconverted). Out: bf16 or f32.
// Block = 256 threads (8 waves). Tile: BM=128, BN=64, BK=32.
// Requires M%128==0, N%64==0, K%32==0 (true for all shapes here).
// ---------------------------------------------------------------------------
static constexpr int BM = 128, BN = 64, BK = 32;
static constexpr int LDA_S = 40;  // padded bf16 stride of As rows (80 B, 16B-mult)
static constexpr int LDB_S = 40;  // padded bf16 stride of Bs rows ([n][k] layout)

template <bool A_BF16, bool OUT_BF16>
__global__ __launch_bounds__(256) void gemm_wmma(
    const void* __restrict__ Av, const unsigned short* __restrict__ Bw,
    const float* __restrict__ bias, void* __restrict__ Cv,
    int M, int N, int K)
{
    __shared__ unsigned short As[BM * LDA_S];
    __shared__ unsigned short Bs[BN * LDB_S];

    const int tid  = threadIdx.x;
    const int wave = tid >> 5;
    const int lane = tid & 31;
    const int l16  = lane & 15;
    const int hl   = lane >> 4;
    const int m0   = blockIdx.x * BM;
    const int n0   = blockIdx.y * BN;

    const f32x8 vzero = {0.f,0.f,0.f,0.f,0.f,0.f,0.f,0.f};
    f32x8 acc[4] = {vzero, vzero, vzero, vzero};

    // cooperative-load coordinates
    const int ar = tid >> 1;            // A tile row 0..127
    const int ac = (tid & 1) * 16;      // A tile col 0 or 16 (16 elems each)
    const int bn = tid & 63;            // B tile col 0..63
    const int bk = (tid >> 6) * 8;      // B tile k 0,8,16,24 (8 elems each)

    const float*          Af = (const float*)Av;
    const unsigned short* Ab = (const unsigned short*)Av;

    for (int k0 = 0; k0 < K; k0 += BK) {
        if (k0 + BK < K) {  // prefetch next A tile chunk -> global_prefetch_b8
            size_t off = (size_t)(m0 + ar) * K + (k0 + BK) + ac;
            __builtin_prefetch((const char*)Av + off * (A_BF16 ? 2 : 4), 0, 1);
        }
        __syncthreads();
        // ---- stage A (convert f32->bf16 if needed) ----
        {
            __align__(16) unsigned short tmp[16];
            if (A_BF16) {
                const unsigned short* src = Ab + (size_t)(m0 + ar) * K + k0 + ac;
                *(uint4*)&tmp[0] = *(const uint4*)(src);
                *(uint4*)&tmp[8] = *(const uint4*)(src + 8);
            } else {
                const float* src = Af + (size_t)(m0 + ar) * K + k0 + ac;
                #pragma unroll
                for (int i = 0; i < 16; i += 4) {
                    float4 f = *(const float4*)(src + i);
                    tmp[i+0] = f2bf(f.x); tmp[i+1] = f2bf(f.y);
                    tmp[i+2] = f2bf(f.z); tmp[i+3] = f2bf(f.w);
                }
            }
            uint4* dst = (uint4*)&As[ar * LDA_S + ac];
            dst[0] = *(uint4*)&tmp[0];
            dst[1] = *(uint4*)&tmp[8];
        }
        // ---- stage B transposed to [n][k] so fragments are contiguous ----
        {
            __align__(16) unsigned short tmp[8];
            #pragma unroll
            for (int i = 0; i < 8; ++i)
                tmp[i] = Bw[(size_t)(k0 + bk + i) * N + n0 + bn];
            *(uint4*)&Bs[bn * LDB_S + bk] = *(uint4*)tmp;
        }
        __syncthreads();

        // ---- WMMA: wave 'wave' owns rows [wave*16, wave*16+16) x 64 cols ----
        FragB a;
        const unsigned short* arow = &As[(wave * 16 + l16) * LDA_S + hl * 8];
        a.u[0] = *(const uint4*)(arow);        // K = 8*hl .. +7
        a.u[1] = *(const uint4*)(arow + 16);   // K = 16+8*hl .. +7
        #pragma unroll
        for (int nt = 0; nt < 4; ++nt) {
            FragB b;
            const unsigned short* brow = &Bs[(nt * 16 + l16) * LDB_S + hl * 8];
            b.u[0] = *(const uint4*)(brow);
            b.u[1] = *(const uint4*)(brow + 16);
            acc[nt] = __builtin_amdgcn_wmma_f32_16x16x32_bf16(
                false, a.v, false, b.v, (short)0, acc[nt], false, false);
        }
    }

    // ---- epilogue: bias + store ----
    #pragma unroll
    for (int nt = 0; nt < 4; ++nt) {
        const int gcol = n0 + nt * 16 + l16;
        const float bv = bias[gcol];
        #pragma unroll
        for (int j = 0; j < 8; ++j) {
            const int grow = m0 + wave * 16 + j + 8 * hl;
            const float val = acc[nt][j] + bv;
            if (OUT_BF16)
                ((unsigned short*)Cv)[(size_t)grow * N + gcol] = f2bf(val);
            else
                ((float*)Cv)[(size_t)grow * N + gcol] = val;
        }
    }
}

// ---------------------------------------------------------------------------
// Region attention: per block one (b,h) and a 128-patch strip (16 per wave).
//   scores = Q(16x64) * K^T(64x32)          -> 4 WMMAs
//   softmax over Nr=32 after *SCALE*p2r+1e-8
//   x = probs(16x32) * V(32x64)             -> 4 WMMAs
// q, kv, x are bf16; p2r is f32.
// ---------------------------------------------------------------------------
__global__ __launch_bounds__(256) void region_attn(
    const unsigned short* __restrict__ qb,
    const unsigned short* __restrict__ kvb,
    const float* __restrict__ p2r,
    unsigned short* __restrict__ xb)
{
    constexpr int NP = 4096, NR = 32, C = 768, D = 64, H = 12;
    constexpr float SCALE = 0.125f;  // 64^-0.5

    __shared__ unsigned short kT[32 * 80];      // K  [r][d], padded
    __shared__ unsigned short vT[64 * 40];      // V^T [d][r], padded
    __shared__ unsigned short pr[8 * 16 * 40];  // per-wave probs [m][r], padded

    const int tid  = threadIdx.x;
    const int bh   = blockIdx.y;
    const int b    = bh / H, h = bh % H;
    const int p0   = blockIdx.x * 128;
    const int wave = tid >> 5, lane = tid & 31;
    const int l16  = lane & 15, hl = lane >> 4;

    // ---- stage K (natural) and V (transposed) for this (b,h) ----
    {
        const int r  = tid >> 3;         // 0..31
        const int d0 = (tid & 7) * 8;    // 0..56
        const unsigned short* row = kvb + (size_t)(b * NR + r) * (2 * C) + h * D + d0;
        *(uint4*)&kT[r * 80 + d0] = *(const uint4*)row;          // K part
        __align__(16) unsigned short vt[8];
        *(uint4*)vt = *(const uint4*)(row + C);                  // V part
        #pragma unroll
        for (int i = 0; i < 8; ++i) vT[(d0 + i) * 40 + r] = vt[i];
    }
    __syncthreads();

    const f32x8 vzero = {0.f,0.f,0.f,0.f,0.f,0.f,0.f,0.f};

    // ---- Q fragments (two K-chunks of 32 over D=64) ----
    const unsigned short* qrow =
        qb + (size_t)(b * NP + p0 + wave * 16 + l16) * C + h * D;
    FragB a0, a1;
    a0.u[0] = *(const uint4*)(qrow + hl * 8);
    a0.u[1] = *(const uint4*)(qrow + 16 + hl * 8);
    a1.u[0] = *(const uint4*)(qrow + 32 + hl * 8);
    a1.u[1] = *(const uint4*)(qrow + 48 + hl * 8);

    // ---- scores: s[nt] covers regions nt*16..+15 ----
    f32x8 s[2] = {vzero, vzero};
    #pragma unroll
    for (int nt = 0; nt < 2; ++nt) {
        const unsigned short* krow = &kT[(nt * 16 + l16) * 80];
        FragB b0, b1;
        b0.u[0] = *(const uint4*)(krow + hl * 8);
        b0.u[1] = *(const uint4*)(krow + 16 + hl * 8);
        b1.u[0] = *(const uint4*)(krow + 32 + hl * 8);
        b1.u[1] = *(const uint4*)(krow + 48 + hl * 8);
        s[nt] = __builtin_amdgcn_wmma_f32_16x16x32_bf16(
            false, a0.v, false, b0.v, (short)0, s[nt], false, false);
        s[nt] = __builtin_amdgcn_wmma_f32_16x16x32_bf16(
            false, a1.v, false, b1.v, (short)0, s[nt], false, false);
    }

    // ---- mask + softmax over 32 regions (row lives across 16 lanes of a half) ----
    float prob[2][8];
    #pragma unroll
    for (int j = 0; j < 8; ++j) {
        const int pp = p0 + wave * 16 + j + 8 * hl;
        const float w0 = p2r[(size_t)(b * NP + pp) * NR + l16];
        const float w1 = p2r[(size_t)(b * NP + pp) * NR + 16 + l16];
        float v0 = s[0][j] * SCALE * w0 + 1e-8f;
        float v1 = s[1][j] * SCALE * w1 + 1e-8f;
        float mx = fmaxf(v0, v1);
        #pragma unroll
        for (int off = 8; off > 0; off >>= 1) mx = fmaxf(mx, __shfl_xor(mx, off, 32));
        float e0 = __expf(v0 - mx), e1 = __expf(v1 - mx);
        float sum = e0 + e1;
        #pragma unroll
        for (int off = 8; off > 0; off >>= 1) sum += __shfl_xor(sum, off, 32);
        const float inv = 1.0f / sum;
        prob[0][j] = e0 * inv;
        prob[1][j] = e1 * inv;
    }

    // ---- re-layout probs (C-layout -> A-fragment layout) via per-wave LDS ----
    unsigned short* pw = &pr[wave * 16 * 40];
    #pragma unroll
    for (int j = 0; j < 8; ++j) {
        const int m = j + 8 * hl;
        pw[m * 40 + l16]      = f2bf(prob[0][j]);
        pw[m * 40 + 16 + l16] = f2bf(prob[1][j]);
    }
    asm volatile("s_wait_dscnt 0x0" ::: "memory");  // wave-local RAW through LDS

    FragB pa;
    const unsigned short* prow = &pw[l16 * 40 + hl * 8];
    pa.u[0] = *(const uint4*)(prow);
    pa.u[1] = *(const uint4*)(prow + 16);

    // ---- x = probs * V ----
    #pragma unroll
    for (int dt = 0; dt < 4; ++dt) {
        const unsigned short* vrow = &vT[(dt * 16 + l16) * 40];
        FragB bv;
        bv.u[0] = *(const uint4*)(vrow + hl * 8);
        bv.u[1] = *(const uint4*)(vrow + 16 + hl * 8);
        f32x8 o = vzero;
        o = __builtin_amdgcn_wmma_f32_16x16x32_bf16(
            false, pa.v, false, bv.v, (short)0, o, false, false);
        #pragma unroll
        for (int j = 0; j < 8; ++j) {
            const int pp = p0 + wave * 16 + j + 8 * hl;
            xb[(size_t)(b * NP + pp) * C + h * D + dt * 16 + l16] = f2bf(o[j]);
        }
    }
}

// ---------------------------------------------------------------------------
// Launch
// ---------------------------------------------------------------------------
extern "C" void kernel_launch(void* const* d_in, const int* in_sizes, int n_in,
                              void* d_out, int out_size, void* d_ws, size_t ws_size,
                              hipStream_t stream) {
    (void)in_sizes; (void)n_in; (void)out_size; (void)ws_size;
    const float* patch  = (const float*)d_in[0];  // (8,4096,768)
    const float* region = (const float*)d_in[1];  // (8,32,768)
    const float* p2r    = (const float*)d_in[2];  // (8,4096,32)
    const float* Wq     = (const float*)d_in[3];
    const float* bq     = (const float*)d_in[4];
    const float* Wkv    = (const float*)d_in[5];
    const float* bkv    = (const float*)d_in[6];
    const float* Wp     = (const float*)d_in[7];
    const float* bp     = (const float*)d_in[8];

    char* ws = (char*)d_ws;
    unsigned short* wq_b  = (unsigned short*)(ws + 0);          //  768* 768*2 = 1179648
    unsigned short* wkv_b = (unsigned short*)(ws + 1179648);    //  768*1536*2 = 2359296
    unsigned short* wp_b  = (unsigned short*)(ws + 3538944);    //  768* 768*2
    unsigned short* q_b   = (unsigned short*)(ws + 4718592);    // 32768*768*2 = 50331648
    unsigned short* kv_b  = (unsigned short*)(ws + 55050240);   //   256*1536*2 = 786432
    unsigned short* x_b   = (unsigned short*)(ws + 55836672);   // 32768*768*2
    // total ~106.2 MB

    // weights -> bf16
    cvt_f32_bf16<<<(768 * 768  + 255) / 256, 256, 0, stream>>>(Wq,  wq_b,  768 * 768);
    cvt_f32_bf16<<<(768 * 1536 + 255) / 256, 256, 0, stream>>>(Wkv, wkv_b, 768 * 1536);
    cvt_f32_bf16<<<(768 * 768  + 255) / 256, 256, 0, stream>>>(Wp,  wp_b,  768 * 768);

    // q = patch @ Wq + bq            (32768 x 768) -> bf16
    gemm_wmma<false, true><<<dim3(32768 / BM, 768 / BN), 256, 0, stream>>>(
        patch, wq_b, bq, q_b, 32768, 768, 768);
    // kv = region @ Wkv + bkv        (256 x 1536) -> bf16
    gemm_wmma<false, true><<<dim3(256 / BM, 1536 / BN), 256, 0, stream>>>(
        region, wkv_b, bkv, kv_b, 256, 1536, 768);
    // attention -> x (bf16)
    region_attn<<<dim3(4096 / 128, 8 * 12), 256, 0, stream>>>(q_b, kv_b, p2r, x_b);
    // out = x @ Wp + bp              (32768 x 768) -> f32
    gemm_wmma<true, false><<<dim3(32768 / BM, 768 / BN), 256, 0, stream>>>(
        x_b, wp_b, bp, d_out, 32768, 768, 768);
}
